// SpatialAttention_24249385353468
// MI455X (gfx1250) — matrix operand order repaired and checked by reading, not compiled, not run on hardware
//
#include <hip/hip_runtime.h>
#include <hip/hip_bf16.h>

typedef __attribute__((ext_vector_type(16))) _Float16 v16h;
typedef __attribute__((ext_vector_type(8)))  _Float16 v8h;
typedef __attribute__((ext_vector_type(8)))  float    v8f;

#define S_LEN 4096
#define C_DIM 512
#define O_DIM 1536
#define D_HEAD 64

// ---------------- WMMA helpers (contiguous k order k = 16*(lane/16) + elem, wave32) ----------------

__device__ __forceinline__ v8f wmma_f16(v16h a, v16h b, v8f c) {
  return __builtin_amdgcn_wmma_f32_16x16x32_f16(false, a, false, b, (short)0, c,
                                                false, false);
}

// A fragment 16x32 f16: lane holds row M = lane%16.
// lanes 0-15: elems 0-15 -> K=0..15
// lanes 16-31: elems 0-15 -> K=16..31
__device__ __forceinline__ v16h load_a_frag(const _Float16* __restrict__ tile, int ld) {
  int lane = threadIdx.x & 31;
  int row = lane & 15;
  int kbase = (lane >> 4) << 4;  // 0 or 16
  const _Float16* p = tile + (long)row * ld;
  v16h a;
#pragma unroll
  for (int j = 0; j < 16; ++j) a[j] = p[kbase + j];
  return a;
}

// B fragment 32x16 f16 from transposed storage Bt[n][k] (column-major B):
// lane holds col N = lane%16; lanes 0-15 K=0..15, lanes 16-31 K=16..31.
// 32 contiguous bytes per lane -> vectorizes to b128 loads.
__device__ __forceinline__ v16h load_bt_frag(const _Float16* __restrict__ tile, int ld) {
  int lane = threadIdx.x & 31;
  int col = lane & 15;
  int kbase = (lane >> 4) << 4;  // 0 or 16
  const _Float16* p = tile + (long)col * ld + kbase;
  v16h b;
#pragma unroll
  for (int j = 0; j < 16; ++j) b[j] = p[j];
  return b;
}

// ---------------- fp32 -> fp16 weight conversion ----------------

__global__ void cvt_kernel(const float* __restrict__ src, _Float16* __restrict__ dst, int n) {
  int i = blockIdx.x * 256 + threadIdx.x;
  if (i < n) dst[i] = (_Float16)src[i];
}

// ---------------- GroupNorm: x[b][c][s] -> hT[b][s][c] (f16) ----------------

__global__ void gn_kernel(const float* __restrict__ x, const float* __restrict__ w,
                          const float* __restrict__ bgn, _Float16* __restrict__ hT) {
  const int CG = 64;                 // channels per group (512/8)
  const int N = CG * S_LEN;          // elements per (b,g)
  int bg = blockIdx.x;               // 0..15
  int b = bg >> 3, g = bg & 7;
  const float* xp = x + ((long)b * C_DIM + g * CG) * S_LEN;

  float s0 = 0.f, s1 = 0.f;
  for (int i = threadIdx.x; i < N; i += 256) {
    float v = xp[i];
    s0 += v; s1 += v * v;
  }
  __shared__ float r0[256], r1[256];
  r0[threadIdx.x] = s0; r1[threadIdx.x] = s1;
  __syncthreads();
  for (int off = 128; off > 0; off >>= 1) {
    if (threadIdx.x < off) {
      r0[threadIdx.x] += r0[threadIdx.x + off];
      r1[threadIdx.x] += r1[threadIdx.x + off];
    }
    __syncthreads();
  }
  float mean = r0[0] / (float)N;
  float var = r1[0] / (float)N - mean * mean;  // biased, matches torch GroupNorm
  float inv = rsqrtf(var + 1e-5f);

  for (int i = threadIdx.x; i < N; i += 256) {
    int cc = i >> 12;          // i / 4096
    int s = i & 4095;
    int c = g * CG + cc;
    float hv = (xp[i] - mean) * inv * w[c] + bgn[c];
    hT[((long)b * S_LEN + s) * C_DIM + c] = (_Float16)hv;
  }
}

// ---------------- QKV GEMM ----------------
// Q,K rows (o < 1024)  -> qkvT[b][s][o]   (contiguous per-lane v8h store)
// V rows  (o >= 1024)  -> vT[b][o-1024][s] (transposed, so attention P*V
//                         B-operand loads become contiguous)
// grid (12, 64, 2), block 256 (8 waves). Macro tile 128(M) x 64(N); wave tile 32x32.

__global__ void qkv_gemm(const _Float16* __restrict__ wA, const _Float16* __restrict__ hT,
                         const float* __restrict__ bias, _Float16* __restrict__ qkvT,
                         _Float16* __restrict__ vT) {
  const int K = C_DIM;
  int b = blockIdx.z;
  int wv = threadIdx.x >> 5;
  int wm = blockIdx.x * 128 + (wv & 3) * 32;
  int wn = blockIdx.y * 64 + (wv >> 2) * 32;
  const _Float16* Bt = hT + (long)b * S_LEN * K;  // [s][c]

  v8f c00 = {}, c01 = {}, c10 = {}, c11 = {};
  for (int k = 0; k < K; k += 32) {
    v16h a0 = load_a_frag(wA + (long)wm * K + k, K);
    v16h a1 = load_a_frag(wA + (long)(wm + 16) * K + k, K);
    v16h b0 = load_bt_frag(Bt + (long)wn * K + k, K);
    v16h b1 = load_bt_frag(Bt + (long)(wn + 16) * K + k, K);
    c00 = wmma_f16(a0, b0, c00);
    c01 = wmma_f16(a0, b1, c01);
    c10 = wmma_f16(a1, b0, c10);
    c11 = wmma_f16(a1, b1, c11);
  }

  int lane = threadIdx.x & 31;
  int mrel = (lane >> 4) << 3;  // 0 or 8
  int nrel = lane & 15;
  auto store_tile = [&](v8f cc, int mt, int nt) {
    int m = mt + mrel;          // multiple of 8
    int s = nt + nrel;
    if (mt >= 1024) {           // V: store transposed [c][s]
#pragma unroll
      for (int r = 0; r < 8; ++r)
        vT[((long)b * C_DIM + (m - 1024) + r) * S_LEN + s] =
            (_Float16)(cc[r] + bias[m + r]);
    } else {                    // Q,K: store [s][o], 16B-aligned v8h
      v8h o;
#pragma unroll
      for (int r = 0; r < 8; ++r) o[r] = (_Float16)(cc[r] + bias[m + r]);
      *(v8h*)(qkvT + ((long)b * S_LEN + s) * O_DIM + m) = o;
    }
  };
  store_tile(c00, wm, wn);
  store_tile(c01, wm, wn + 16);
  store_tile(c10, wm + 16, wn);
  store_tile(c11, wm + 16, wn + 16);
}

// ---------------- Flash attention: grid (64, 8, 2), block 128 (4 waves) ----------------
// Each wave: 16 query rows x full dh=64. Online softmax in f32.

__global__ void attn_kernel(const _Float16* __restrict__ qkvT,
                            const _Float16* __restrict__ vT,
                            _Float16* __restrict__ outT) {
  int b = blockIdx.z, hh = blockIdx.y;
  int wv = threadIdx.x >> 5;
  int lane = threadIdx.x & 31;
  int qrow = blockIdx.x * 64 + wv * 16;

  const _Float16* base = qkvT + (long)b * S_LEN * O_DIM;
  const _Float16* Q = base + hh * D_HEAD;                         // [s][o], o 0..511
  const _Float16* Km = base + 512 + hh * D_HEAD;                  // [s][o], o 512..1023
  const _Float16* Vt = vT + ((long)b * C_DIM + hh * D_HEAD) * S_LEN;  // [d][s]
  const float scale = 0.125f;                                     // 64^-0.5

  __shared__ _Float16 plds[4][16 * 72];  // per-wave P staging, padded
  _Float16* pw = &plds[wv][0];

  // Q fragments: rows qrow..qrow+15, d 0..31 and 32..63
  v16h qa0 = load_a_frag(Q + (long)qrow * O_DIM, O_DIM);
  v16h qa1 = load_a_frag(Q + (long)qrow * O_DIM + 32, O_DIM);

  v8f o[4] = {v8f{}, v8f{}, v8f{}, v8f{}};
  float mrow[8], lrow[8];
#pragma unroll
  for (int r = 0; r < 8; ++r) { mrow[r] = -1e30f; lrow[r] = 0.f; }

  int prow = (lane >> 4) << 3;  // 0 or 8 (row half of C/D layout)
  int pcol = lane & 15;

  for (int kt = 0; kt < S_LEN; kt += 64) {
    // ---- scores S = scale * Q K^T (16 x 64), 8 WMMAs ----
    v8f sc[4];
#pragma unroll
    for (int t = 0; t < 4; ++t) {
      v16h kb0 = load_bt_frag(Km + (long)(kt + t * 16) * O_DIM, O_DIM);
      v16h kb1 = load_bt_frag(Km + (long)(kt + t * 16) * O_DIM + 32, O_DIM);
      v8f acc = {};
      acc = wmma_f16(qa0, kb0, acc);
      acc = wmma_f16(qa1, kb1, acc);
      sc[t] = acc;
    }
    // ---- row max over 64 keys ----
    float rmax[8];
#pragma unroll
    for (int r = 0; r < 8; ++r) {
      float v = -1e30f;
#pragma unroll
      for (int t = 0; t < 4; ++t) { sc[t][r] *= scale; v = fmaxf(v, sc[t][r]); }
      rmax[r] = v;
    }
#pragma unroll
    for (int off = 1; off < 16; off <<= 1)
#pragma unroll
      for (int r = 0; r < 8; ++r) rmax[r] = fmaxf(rmax[r], __shfl_xor(rmax[r], off, 16));

    float alpha[8];
#pragma unroll
    for (int r = 0; r < 8; ++r) {
      float mn = fmaxf(mrow[r], rmax[r]);
      alpha[r] = __expf(mrow[r] - mn);
      mrow[r] = mn;
    }
    // ---- P = exp(S - m), row sums, stage P to LDS as f16 ----
    float rsum[8];
#pragma unroll
    for (int r = 0; r < 8; ++r) rsum[r] = 0.f;
#pragma unroll
    for (int t = 0; t < 4; ++t) {
#pragma unroll
      for (int r = 0; r < 8; ++r) {
        float p = __expf(sc[t][r] - mrow[r]);
        rsum[r] += p;
        pw[(prow + r) * 72 + t * 16 + pcol] = (_Float16)p;
      }
    }
#pragma unroll
    for (int off = 1; off < 16; off <<= 1)
#pragma unroll
      for (int r = 0; r < 8; ++r) rsum[r] += __shfl_xor(rsum[r], off, 16);
#pragma unroll
    for (int r = 0; r < 8; ++r) lrow[r] = lrow[r] * alpha[r] + rsum[r];
#pragma unroll
    for (int t = 0; t < 4; ++t)
#pragma unroll
      for (int r = 0; r < 8; ++r) o[t][r] *= alpha[r];

    asm volatile("s_wait_dscnt 0" ::: "memory");  // LDS RAW fence (same wave)

    // ---- O += P V (16 x 64), 8 WMMAs; V B-operand from transposed vT ----
    v16h pa0 = load_a_frag(pw, 72);        // keys 0..31 of tile
    v16h pa1 = load_a_frag(pw + 32, 72);   // keys 32..63
#pragma unroll
    for (int t = 0; t < 4; ++t) {
      v16h vb0 = load_bt_frag(Vt + (long)(t * 16) * S_LEN + kt, S_LEN);
      v16h vb1 = load_bt_frag(Vt + (long)(t * 16) * S_LEN + kt + 32, S_LEN);
      o[t] = wmma_f16(pa0, vb0, o[t]);
      o[t] = wmma_f16(pa1, vb1, o[t]);
    }
  }

  // ---- epilogue: normalize and store to outT[b][s][c] (f16) ----
#pragma unroll
  for (int t = 0; t < 4; ++t) {
#pragma unroll
    for (int r = 0; r < 8; ++r) {
      int s = qrow + prow + r;
      int c = hh * D_HEAD + t * 16 + pcol;
      outT[((long)b * S_LEN + s) * C_DIM + c] = (_Float16)(o[t][r] / lrow[r]);
    }
  }
}

// ---------------- Proj GEMM + bias + residual -> out[b][o][s] fp32 ----------------
// grid (4, 64, 2), block 256 (8 waves).

__global__ void proj_gemm(const _Float16* __restrict__ wA, const _Float16* __restrict__ aoT,
                          const float* __restrict__ bias, const float* __restrict__ xin,
                          float* __restrict__ outp) {
  const int K = C_DIM;
  int b = blockIdx.z;
  int wv = threadIdx.x >> 5;
  int wm = blockIdx.x * 128 + (wv & 3) * 32;
  int wn = blockIdx.y * 64 + (wv >> 2) * 32;
  const _Float16* Bt = aoT + (long)b * S_LEN * K;  // [s][c]

  v8f c00 = {}, c01 = {}, c10 = {}, c11 = {};
  for (int k = 0; k < K; k += 32) {
    v16h a0 = load_a_frag(wA + (long)wm * K + k, K);
    v16h a1 = load_a_frag(wA + (long)(wm + 16) * K + k, K);
    v16h b0 = load_bt_frag(Bt + (long)wn * K + k, K);
    v16h b1 = load_bt_frag(Bt + (long)(wn + 16) * K + k, K);
    c00 = wmma_f16(a0, b0, c00);
    c01 = wmma_f16(a0, b1, c01);
    c10 = wmma_f16(a1, b0, c10);
    c11 = wmma_f16(a1, b1, c11);
  }

  int lane = threadIdx.x & 31;
  int mrel = (lane >> 4) << 3;
  int nrel = lane & 15;
  auto store_tile = [&](v8f cc, int mt, int nt) {
    int m = mt + mrel;
    int s = nt + nrel;
#pragma unroll
    for (int r = 0; r < 8; ++r) {
      long idx = ((long)b * C_DIM + m + r) * S_LEN + s;
      outp[idx] = cc[r] + xin[idx] + bias[m + r];
    }
  };
  store_tile(c00, wm, wn);
  store_tile(c01, wm, wn + 16);
  store_tile(c10, wm + 16, wn);
  store_tile(c11, wm + 16, wn + 16);
}

// ---------------- host launch ----------------

extern "C" void kernel_launch(void* const* d_in, const int* in_sizes, int n_in,
                              void* d_out, int out_size, void* d_ws, size_t ws_size,
                              hipStream_t stream) {
  (void)in_sizes; (void)n_in; (void)out_size; (void)ws_size;
  const float* x      = (const float*)d_in[0];
  const float* gn_w   = (const float*)d_in[1];
  const float* gn_b   = (const float*)d_in[2];
  const float* qkv_w  = (const float*)d_in[3];
  const float* qkv_b  = (const float*)d_in[4];
  const float* proj_w = (const float*)d_in[5];
  const float* proj_b = (const float*)d_in[6];
  float* out = (float*)d_out;

  char* ws = (char*)d_ws;
  _Float16* hT = (_Float16*)ws;   ws += (size_t)2 * S_LEN * C_DIM * sizeof(_Float16);   // 8 MiB
  _Float16* wq = (_Float16*)ws;   ws += (size_t)O_DIM * C_DIM * sizeof(_Float16);       // 1.5 MiB
  _Float16* wp = (_Float16*)ws;   ws += (size_t)C_DIM * C_DIM * sizeof(_Float16);       // 0.5 MiB
  _Float16* qkvT = (_Float16*)ws; ws += (size_t)2 * S_LEN * O_DIM * sizeof(_Float16);   // 24 MiB
  _Float16* vT = (_Float16*)ws;   ws += (size_t)2 * C_DIM * S_LEN * sizeof(_Float16);   // 8 MiB
  _Float16* aoT = (_Float16*)ws;  // 8 MiB

  cvt_kernel<<<(O_DIM * C_DIM + 255) / 256, 256, 0, stream>>>(qkv_w, wq, O_DIM * C_DIM);
  cvt_kernel<<<(C_DIM * C_DIM + 255) / 256, 256, 0, stream>>>(proj_w, wp, C_DIM * C_DIM);
  gn_kernel<<<16, 256, 0, stream>>>(x, gn_w, gn_b, hT);
  qkv_gemm<<<dim3(12, 64, 2), 256, 0, stream>>>(wq, hT, qkv_b, qkvT, vT);
  attn_kernel<<<dim3(64, 8, 2), 128, 0, stream>>>(qkvT, vT, aoT);
  proj_gemm<<<dim3(4, 64, 2), 256, 0, stream>>>(wp, aoT, proj_b, x, out);
}
